// ExpertLayer_21981642621300
// MI455X (gfx1250) — matrix operand order repaired and checked by reading, not compiled, not run on hardware
//
#include <hip/hip_runtime.h>

// ---------------------------------------------------------------------------
// Switch-Transformer top-1 MoE forward for MI455X (gfx1250, wave32).
// Heavy math runs on v_wmma_f32_16x16x32_bf16 using a bf16 hi/lo split of the
// fp32 operands (3 WMMAs per tile) for ~fp32 accuracy with 8x the per-instr
// FLOPs of the f32 WMMA path. Roofline: 137 GFLOP, ~600MB HBM traffic.
// ---------------------------------------------------------------------------

typedef __attribute__((ext_vector_type(16))) __bf16 v16bf;
typedef __attribute__((ext_vector_type(8)))  float  v8f;

#define MBLK 128
#define NBLK 128
#define KBLK 32
#define WROW 20   // padded uint32 words per LDS row (16 data + 4 pad); 80B rows

// Split fp32 f0,f1 into packed bf16 hi-word and lo-word (truncation split;
// hi+lo carries ~16 mantissa bits => product error ~2^-16 relative).
__device__ __forceinline__ void split2(float f0, float f1,
                                       unsigned& hw, unsigned& lw) {
  unsigned u0 = __float_as_uint(f0), u1 = __float_as_uint(f1);
  unsigned h0 = u0 & 0xFFFF0000u,    h1 = u1 & 0xFFFF0000u;
  hw = (h0 >> 16) | h1;
  float r0 = f0 - __uint_as_float(h0);
  float r1 = f1 - __uint_as_float(h1);
  lw = ((__float_as_uint(r0) & 0xFFFF0000u) >> 16) |
       (__float_as_uint(r1) & 0xFFFF0000u);
}

__device__ __forceinline__ float4 ld4(const float* p) {
  return *(const float4*)p;
}

struct U8 { uint4 a, b; };

// Fragment gather: two 16B LDS chunks -> v16bf (forces 2 x ds_load_b128).
__device__ __forceinline__ v16bf load_frag(const unsigned* p0,
                                           const unsigned* p1) {
  U8 t;
  t.a = *(const uint4*)p0;
  t.b = *(const uint4*)p1;
  return __builtin_bit_cast(v16bf, t);
}

// ---------------------------------------------------------------------------
// Grouped GEMM: for each expert e (blockIdx.z):
//   C[e] = act(A[e][M,K] @ B[e][K,N] + bias[e][N])
// Block tile 128x128, 8 waves of 32, wave tile 64x32, K-step 32.
// ---------------------------------------------------------------------------
template<bool RELU>
__global__ __launch_bounds__(256)
void gemm_moe(const float* __restrict__ A, const float* __restrict__ B,
              const float* __restrict__ bias, float* __restrict__ C,
              int M, int N, int K) {
  __shared__ __align__(16) unsigned Ah[MBLK * WROW];
  __shared__ __align__(16) unsigned Al[MBLK * WROW];
  __shared__ __align__(16) unsigned Bh[NBLK * WROW];
  __shared__ __align__(16) unsigned Bl[NBLK * WROW];

  const int e  = blockIdx.z;
  const int m0 = blockIdx.y * MBLK;
  const int n0 = blockIdx.x * NBLK;
  const float* Ae = A + (size_t)e * M * K;
  const float* Be = B + (size_t)e * K * N;
  const float* be = bias + (size_t)e * N;
  float*       Ce = C + (size_t)e * M * N;

  const int tid  = threadIdx.x;
  const int lane = tid & 31;
  const int wave = tid >> 5;
  const int wm = (wave & 1) * 64;   // wave M offset in block tile
  const int wn = (wave >> 1) * 32;  // wave N offset in block tile

  v8f acc[4][2];
#pragma unroll
  for (int mi = 0; mi < 4; ++mi)
#pragma unroll
    for (int ni = 0; ni < 2; ++ni) {
      v8f z = {0.f, 0.f, 0.f, 0.f, 0.f, 0.f, 0.f, 0.f};
      acc[mi][ni] = z;
    }

  // Global-load thread mapping (coalesced along the contiguous dimension)
  const int arow = tid >> 3;         // 0..31  (A row within 32-row pass)
  const int acol = (tid & 7) << 2;   // 0..28  (A k offset, float4)
  const int bn   = (tid & 31) << 2;  // 0..124 (B n offset, float4)
  const int bk   = (tid >> 5) << 1;  // 0..14  (B k-pair within 16-row pass)

  float4 ra0, ra1, ra2, ra3;          // A tile: 4 passes of 32 rows
  float4 rb00, rb01, rb10, rb11;      // B tile: 2 passes x (k, k+1)

  const int KT = K / KBLK;

#define LOAD_TILE(kt_)                                                        \
  {                                                                           \
    const int k0 = (kt_) * KBLK;                                              \
    ra0 = ld4(Ae + (size_t)(m0 + arow      ) * K + k0 + acol);                \
    ra1 = ld4(Ae + (size_t)(m0 + arow + 32 ) * K + k0 + acol);                \
    ra2 = ld4(Ae + (size_t)(m0 + arow + 64 ) * K + k0 + acol);                \
    ra3 = ld4(Ae + (size_t)(m0 + arow + 96 ) * K + k0 + acol);                \
    rb00 = ld4(Be + (size_t)(k0 + bk     ) * N + n0 + bn);                    \
    rb01 = ld4(Be + (size_t)(k0 + bk + 1 ) * N + n0 + bn);                    \
    rb10 = ld4(Be + (size_t)(k0 + bk + 16) * N + n0 + bn);                    \
    rb11 = ld4(Be + (size_t)(k0 + bk + 17) * N + n0 + bn);                    \
  }

#define STORE_TILE()                                                          \
  {                                                                           \
    unsigned h0, l0, h1, l1;                                                  \
    int wb = arow * WROW + ((tid & 7) << 1);                                  \
    split2(ra0.x, ra0.y, h0, l0); split2(ra0.z, ra0.w, h1, l1);               \
    Ah[wb] = h0; Ah[wb + 1] = h1; Al[wb] = l0; Al[wb + 1] = l1;               \
    wb += 32 * WROW;                                                          \
    split2(ra1.x, ra1.y, h0, l0); split2(ra1.z, ra1.w, h1, l1);               \
    Ah[wb] = h0; Ah[wb + 1] = h1; Al[wb] = l0; Al[wb + 1] = l1;               \
    wb += 32 * WROW;                                                          \
    split2(ra2.x, ra2.y, h0, l0); split2(ra2.z, ra2.w, h1, l1);               \
    Ah[wb] = h0; Ah[wb + 1] = h1; Al[wb] = l0; Al[wb + 1] = l1;               \
    wb += 32 * WROW;                                                          \
    split2(ra3.x, ra3.y, h0, l0); split2(ra3.z, ra3.w, h1, l1);               \
    Ah[wb] = h0; Ah[wb + 1] = h1; Al[wb] = l0; Al[wb + 1] = l1;               \
    unsigned hw, lw;                                                          \
    const int w0 = bk >> 1;        /* word idx for K pair (bk,bk+1)   */      \
    const int w1 = 8 + (bk >> 1);  /* word idx for K pair (16+bk,..)  */      \
    split2(rb00.x, rb01.x, hw, lw); Bh[(bn+0)*WROW + w0] = hw; Bl[(bn+0)*WROW + w0] = lw; \
    split2(rb00.y, rb01.y, hw, lw); Bh[(bn+1)*WROW + w0] = hw; Bl[(bn+1)*WROW + w0] = lw; \
    split2(rb00.z, rb01.z, hw, lw); Bh[(bn+2)*WROW + w0] = hw; Bl[(bn+2)*WROW + w0] = lw; \
    split2(rb00.w, rb01.w, hw, lw); Bh[(bn+3)*WROW + w0] = hw; Bl[(bn+3)*WROW + w0] = lw; \
    split2(rb10.x, rb11.x, hw, lw); Bh[(bn+0)*WROW + w1] = hw; Bl[(bn+0)*WROW + w1] = lw; \
    split2(rb10.y, rb11.y, hw, lw); Bh[(bn+1)*WROW + w1] = hw; Bl[(bn+1)*WROW + w1] = lw; \
    split2(rb10.z, rb11.z, hw, lw); Bh[(bn+2)*WROW + w1] = hw; Bl[(bn+2)*WROW + w1] = lw; \
    split2(rb10.w, rb11.w, hw, lw); Bh[(bn+3)*WROW + w1] = hw; Bl[(bn+3)*WROW + w1] = lw; \
  }

  LOAD_TILE(0);
  STORE_TILE();
  __syncthreads();

  const int kg = lane >> 4;      // half-wave K group
  const int lm = lane & 15;      // lane row/col within 16

  for (int kt = 0; kt < KT; ++kt) {
    if (kt + 1 < KT) LOAD_TILE(kt + 1);  // register prefetch next tile

    v16bf ah[4], al[4], bh[2], bl[2];
#pragma unroll
    for (int mi = 0; mi < 4; ++mi) {
      // A 16x32 bf16 layout: chunk0 = words[kg*4 .. +3], chunk1 = +8
      const int base = (wm + mi * 16 + lm) * WROW + (kg << 2);
      ah[mi] = load_frag(&Ah[base], &Ah[base + 8]);
      al[mi] = load_frag(&Al[base], &Al[base + 8]);
    }
#pragma unroll
    for (int ni = 0; ni < 2; ++ni) {
      // B 32x16 bf16 layout: chunk0 = words[kg*4 .. +3], chunk1 = +8
      const int base = (wn + ni * 16 + lm) * WROW + (kg << 2);
      bh[ni] = load_frag(&Bh[base], &Bh[base + 8]);
      bl[ni] = load_frag(&Bl[base], &Bl[base + 8]);
    }

#pragma unroll
    for (int mi = 0; mi < 4; ++mi)
#pragma unroll
      for (int ni = 0; ni < 2; ++ni) {
        acc[mi][ni] = __builtin_amdgcn_wmma_f32_16x16x32_bf16(
            false, ah[mi], false, bh[ni], (short)0, acc[mi][ni], false, false);
        acc[mi][ni] = __builtin_amdgcn_wmma_f32_16x16x32_bf16(
            false, ah[mi], false, bl[ni], (short)0, acc[mi][ni], false, false);
        acc[mi][ni] = __builtin_amdgcn_wmma_f32_16x16x32_bf16(
            false, al[mi], false, bh[ni], (short)0, acc[mi][ni], false, false);
      }

    __syncthreads();
    if (kt + 1 < KT) STORE_TILE();
    __syncthreads();
  }

  // Epilogue: C 16x16 f32 layout: VGPR g -> row g + 8*(lane>>4), col lane&15
#pragma unroll
  for (int ni = 0; ni < 2; ++ni) {
    const int col = n0 + wn + ni * 16 + lm;
    const float bv = be[col];
#pragma unroll
    for (int mi = 0; mi < 4; ++mi) {
      const int rbase = m0 + wm + mi * 16 + (kg << 3);
#pragma unroll
      for (int g = 0; g < 8; ++g) {
        float v = acc[mi][ni][g] + bv;
        if (RELU) v = v > 0.f ? v : 0.f;
        Ce[(size_t)(rbase + g) * N + col] = v;
      }
    }
  }
#undef LOAD_TILE
#undef STORE_TILE
}

// ---------------------------------------------------------------------------
// Router: one wave per token. logits = x[t] @ Wg, softmax, top-1, gate,
// plus atomically-accumulated stats for aux/z losses.
// ---------------------------------------------------------------------------
__global__ __launch_bounds__(256)
void router_kernel(const float* __restrict__ x, const float* __restrict__ Wg,
                   int* __restrict__ eidx, float* __restrict__ gate,
                   float* __restrict__ psum, float* __restrict__ zsum,
                   int* __restrict__ counts, int T, int D) {
  const int wave = threadIdx.x >> 5;
  const int lane = threadIdx.x & 31;
  const int t = blockIdx.x * 8 + wave;
  if (t >= T) return;

  float acc[8] = {0.f, 0.f, 0.f, 0.f, 0.f, 0.f, 0.f, 0.f};
  const float* xr = x + (size_t)t * D;
  for (int d = lane; d < D; d += 32) {
    const float xv = xr[d];
    const float* wr = Wg + (size_t)d * 8;
#pragma unroll
    for (int e = 0; e < 8; ++e) acc[e] += xv * wr[e];
  }
#pragma unroll
  for (int e = 0; e < 8; ++e)
#pragma unroll
    for (int off = 16; off > 0; off >>= 1)
      acc[e] += __shfl_down(acc[e], off, 32);

  if (lane == 0) {
    float m = acc[0];
#pragma unroll
    for (int e = 1; e < 8; ++e) m = acc[e] > m ? acc[e] : m;
    float p[8], s = 0.f;
#pragma unroll
    for (int e = 0; e < 8; ++e) { p[e] = __expf(acc[e] - m); s += p[e]; }
    const float inv = 1.f / s;
    int best = 0; float bv = acc[0];
#pragma unroll
    for (int e = 1; e < 8; ++e)
      if (acc[e] > bv) { bv = acc[e]; best = e; }  // first max wins
    eidx[t] = best;
    gate[t] = p[best] * inv;
    const float lse = m + __logf(s);
    atomicAdd(zsum, lse * lse);
#pragma unroll
    for (int e = 0; e < 8; ++e) atomicAdd(&psum[e], p[e] * inv);
    atomicAdd(&counts[best], 1);
  }
}

// ---------------------------------------------------------------------------
// Dispatch positions: sequential cumsum over tokens. One wave: lane e is the
// counter for expert e; token ids broadcast via shuffle. O(T), negligible.
// ---------------------------------------------------------------------------
__global__ void pos_kernel(const int* __restrict__ eidx,
                           const float* __restrict__ gate,
                           int* __restrict__ posc, int* __restrict__ keepf,
                           float* __restrict__ gatek, int T, int cap) {
  const int lane = threadIdx.x;  // blockDim == 32
  int cnt = 0;
  for (int t0 = 0; t0 < T; t0 += 32) {
    const int ev = eidx[t0 + lane];
    for (int j = 0; j < 32; ++j) {
      const int e = __shfl(ev, j, 32);
      if (lane == e) {
        const int p = cnt++;
        const int keep = (p < cap) ? 1 : 0;
        const int t = t0 + j;
        posc[t] = keep ? p : (cap - 1);
        keepf[t] = keep;
        gatek[t] = keep ? gate[t] : 0.f;
      }
    }
  }
}

__global__ __launch_bounds__(256)
void scatter_kernel(const float* __restrict__ x, const int* __restrict__ eidx,
                    const int* __restrict__ posc, const int* __restrict__ keepf,
                    float* __restrict__ buf, int D, int cap) {
  const int t = blockIdx.x;
  if (!keepf[t]) return;
  const int e = eidx[t], p = posc[t];
  const float4* src = (const float4*)(x + (size_t)t * D);
  float4* dst = (float4*)(buf + ((size_t)e * cap + p) * D);
  for (int i = threadIdx.x; i < D / 4; i += 256) dst[i] = src[i];
}

__global__ __launch_bounds__(256)
void combine_kernel(const float* __restrict__ ob, const int* __restrict__ eidx,
                    const int* __restrict__ posc, const float* __restrict__ gatek,
                    float* __restrict__ y, int D, int cap) {
  const int t = blockIdx.x;
  const int e = eidx[t], p = posc[t];
  const float g = gatek[t];
  const float4* src = (const float4*)(ob + ((size_t)e * cap + p) * D);
  float4* dst = (float4*)(y + (size_t)t * D);
  for (int i = threadIdx.x; i < D / 4; i += 256) {
    float4 v = src[i];
    v.x *= g; v.y *= g; v.z *= g; v.w *= g;
    dst[i] = v;
  }
}

__global__ void finalize_kernel(const int* __restrict__ counts,
                                const float* __restrict__ psum,
                                const float* __restrict__ zsum,
                                float* __restrict__ out, int T, int E) {
  if (threadIdx.x == 0 && blockIdx.x == 0) {
    float aux = 0.f;
    for (int e = 0; e < E; ++e)
      aux += ((float)counts[e] / (float)T) * (psum[e] / (float)T);
    out[0] = (float)E * aux;
    out[1] = zsum[0] / (float)T;
  }
}

__global__ void zero_f(float* __restrict__ p, size_t n) {
  size_t i = (size_t)blockIdx.x * blockDim.x + threadIdx.x;
  const size_t stride = (size_t)gridDim.x * blockDim.x;
  for (; i < n; i += stride) p[i] = 0.f;
}

// ---------------------------------------------------------------------------
extern "C" void kernel_launch(void* const* d_in, const int* in_sizes, int n_in,
                              void* d_out, int out_size, void* d_ws, size_t ws_size,
                              hipStream_t stream) {
  (void)in_sizes; (void)n_in; (void)out_size; (void)ws_size;
  const float* x  = (const float*)d_in[0];
  const float* Wg = (const float*)d_in[1];
  const float* W1 = (const float*)d_in[2];
  const float* b1 = (const float*)d_in[3];
  const float* W2 = (const float*)d_in[4];
  const float* b2 = (const float*)d_in[5];
  float* y = (float*)d_out;

  constexpr int Bn = 2, Sn = 2048, Dn = 1024, En = 8, Fn = 4096;
  const int T = Bn * Sn;          // 4096 tokens
  const int cap = (2 * T) / En;   // 1024 per-expert capacity

  // Workspace layout: buf[E,cap,D] (32MB, reused as OB) | H[E,cap,F] (128MB) | meta
  const size_t bufElems = (size_t)En * cap * Dn;  // 8M floats
  const size_t hElems   = (size_t)En * cap * Fn;  // 32M floats
  float* buf = (float*)d_ws;
  float* H   = buf + bufElems;
  char* mp   = (char*)(H + hElems);
  int*   eidx  = (int*)mp;   mp += (size_t)T * sizeof(int);
  int*   posc  = (int*)mp;   mp += (size_t)T * sizeof(int);
  int*   keepf = (int*)mp;   mp += (size_t)T * sizeof(int);
  float* gate  = (float*)mp; mp += (size_t)T * sizeof(float);
  float* gatek = (float*)mp; mp += (size_t)T * sizeof(float);
  int*   counts = (int*)mp;  mp += 8 * sizeof(int);
  float* psum   = (float*)mp; mp += 8 * sizeof(float);
  float* zsum   = (float*)mp; mp += sizeof(float);

  // Zero dispatch buffers + loss accumulators (graph-capture safe: kernels only)
  zero_f<<<2048, 256, 0, stream>>>(buf, bufElems);
  zero_f<<<1, 32, 0, stream>>>((float*)counts, 17);

  router_kernel<<<T / 8, 256, 0, stream>>>(x, Wg, eidx, gate, psum, zsum,
                                           counts, T, Dn);
  pos_kernel<<<1, 32, 0, stream>>>(eidx, gate, posc, keepf, gatek, T, cap);
  scatter_kernel<<<T, 256, 0, stream>>>(x, eidx, posc, keepf, buf, Dn, cap);

  // Expert FFN: H = relu(buf @ W1 + b1); OB = H @ W2 + b2 (OB reuses buf region)
  dim3 g1(Fn / NBLK, cap / MBLK, En);  // (32, 8, 8), expert-major for L2 reuse
  gemm_moe<true><<<g1, 256, 0, stream>>>(buf, W1, b1, H, cap, Fn, Dn);
  dim3 g2(Dn / NBLK, cap / MBLK, En);  // (8, 8, 8)
  gemm_moe<false><<<g2, 256, 0, stream>>>(H, W2, b2, buf, cap, Dn, Fn);

  combine_kernel<<<T, 256, 0, stream>>>(buf, eidx, posc, gatek, y, Dn, cap);
  finalize_kernel<<<1, 1, 0, stream>>>(counts, psum, zsum,
                                       y + (size_t)T * Dn, T, En);
}
